// CyclicTiSASRec_65274912964707
// MI455X (gfx1250) — matrix operand order repaired and checked by reading, not compiled, not run on hardware
//
#include <hip/hip_runtime.h>
#include <hip/hip_bf16.h>

#define BB   8
#define TT   256
#define HH   128
#define NHH  2
#define DHH  64
#define LL   2
#define NTOK (BB*TT)
#define NTH  (NTOK*HH)
#define NEGV (-1e9f)
#define TWO_PI 6.283185307179586f

typedef __attribute__((ext_vector_type(16))) _Float16 v16h;
typedef __attribute__((ext_vector_type(8)))  float    v8f;

// Optional gfx1250 async global->LDS path (guarded; falls back to plain loads)
#if defined(__has_builtin)
#  if __has_builtin(__builtin_amdgcn_global_load_async_to_lds_b128) && \
      __has_builtin(__builtin_amdgcn_s_wait_asynccnt)
#    define USE_ASYNC_LDS 1
#  endif
#endif
#ifndef USE_ASYNC_LDS
#  define USE_ASYNC_LDS 0
#endif

#if USE_ASYNC_LDS
typedef int v4i_t __attribute__((ext_vector_type(4)));
typedef __attribute__((address_space(1))) v4i_t* g_v4i_p;   // global int4*
typedef __attribute__((address_space(3))) v4i_t* s_v4i_p;   // LDS int4*
#endif

// ---------------------------------------------------------------------------
// WMMA GEMM, one wave computes a 16(M) x 64(N) strip: A fragment loaded once
// per 32-wide K step and reused across 4 v_wmma_f32_16x16x32_f16 issues.
//   BTRANS=1: Bmat(k,n) = W[n*ldw + k]  (X @ W^T)     BTRANS=0: W[k*ldw + n]
//   ACT: 0 none, 1 relu, 2 exact gelu
//   OMODE=0: Out[m*ldo+n] (+ optional residual extra[m*ldo+n])
//   OMODE=1: head-split (B,NH,T,DH) write, extra = abs-pos table (T,H)
// ---------------------------------------------------------------------------
template<int BTRANS, int ACT, int OMODE>
__global__ __launch_bounds__(32)
void wmma_gemm_k(const float* __restrict__ X, const float* __restrict__ W,
                 const float* __restrict__ bias, const float* __restrict__ extra,
                 float* __restrict__ Out,
                 int M, int N, int K, int ldx, int ldw, int ldo,
                 long bsX, long bsW, long bsO)
{
    const int lane = threadIdx.x;
    const int tn = blockIdx.x, tm = blockIdx.y, bz = blockIdx.z;
    const float* Xb = X + (long)bz * bsX;
    const float* Wb = W + (long)bz * bsW;
    float*       Ob = Out + (long)bz * bsO;

    const int l15 = lane & 15;
    const int hi  = lane >> 4;            // 0 | 1
    const int mrow = tm * 16 + l15;       // A row for this lane
    const int kbA = hi * 8;               // A-frag K offset for hi lanes
    const int kbB = hi * 16;              // unused: B frag uses the same hi*8 K offset as A

    v8f acc[4];
#pragma unroll
    for (int nt = 0; nt < 4; ++nt) acc[nt] = (v8f){0.f,0.f,0.f,0.f,0.f,0.f,0.f,0.f};

    for (int k0 = 0; k0 < K; k0 += 32) {
        v16h afrag;
#pragma unroll
        for (int e = 0; e < 16; ++e) {    // A 16x32 f16 layout (ISA 7.12.2)
            int ka = k0 + ((e >> 3) << 4) + kbA + (e & 7);
            afrag[e] = (_Float16)Xb[(long)mrow * ldx + ka];
        }
        if (k0 + 32 < K)                  // global_prefetch_b8 of next A chunk
            __builtin_prefetch(&Xb[(long)mrow * ldx + k0 + 32], 0, 3);
#pragma unroll
        for (int nt = 0; nt < 4; ++nt) {
            const int ncol = tn * 64 + nt * 16 + l15;
            v16h bfrag;
#pragma unroll
            for (int e = 0; e < 16; ++e) {  // B 32x16 f16 layout
                int kb = k0 + ((e >> 3) << 4) + hi * 8 + (e & 7);
                float wv = BTRANS ? Wb[(long)ncol * ldw + kb]
                                  : Wb[(long)kb * ldw + ncol];
                bfrag[e] = (_Float16)wv;
            }
            acc[nt] = __builtin_amdgcn_wmma_f32_16x16x32_f16(
                          false, afrag, false, bfrag, (short)0, acc[nt], false, false);
        }
    }

#pragma unroll
    for (int nt = 0; nt < 4; ++nt) {
        const int n = tn * 64 + nt * 16 + l15;
#pragma unroll
        for (int j = 0; j < 8; ++j) {     // D layout: M=j+8*hi, N=l15
            int m = tm * 16 + j + hi * 8;
            float val = acc[nt][j];
            if (bias) val += bias[n];
            if (ACT == 1)      val = fmaxf(val, 0.f);
            else if (ACT == 2) val = 0.5f * val * (1.f + erff(val * 0.7071067811865476f));
            if (OMODE == 0) {
                if (extra) val += extra[(long)m * ldo + n];
                Ob[(long)m * ldo + n] = val;
            } else {
                int b = m / TT, t = m % TT;
                int h = n / DHH, d = n % DHH;
                if (extra) val += extra[(long)t * HH + n];   // fold abs_pos_K/V
                Ob[(((long)b * NHH + h) * TT + t) * DHH + d] = val;
            }
        }
    }
}

// ---------------------------------------------------------------------------
__global__ __launch_bounds__(HH)
void ln_k(const float* __restrict__ in, const float* __restrict__ g,
          const float* __restrict__ b, float* __restrict__ out)
{
    __shared__ float red[HH];
    int row = blockIdx.x, tid = threadIdx.x;
    float x = in[(long)row * HH + tid];
    red[tid] = x; __syncthreads();
    for (int s = HH/2; s > 0; s >>= 1) { if (tid < s) red[tid] += red[tid+s]; __syncthreads(); }
    float mu = red[0] * (1.0f / HH);
    __syncthreads();
    float dx = x - mu;
    red[tid] = dx * dx; __syncthreads();
    for (int s = HH/2; s > 0; s >>= 1) { if (tid < s) red[tid] += red[tid+s]; __syncthreads(); }
    float var = red[0] * (1.0f / HH);
    out[(long)row * HH + tid] = dx * rsqrtf(var + 1e-8f) * g[tid] + b[tid];
}

__global__ __launch_bounds__(HH)
void embed_k(const int* __restrict__ logs, const float* __restrict__ tseq,
             const float* __restrict__ item_emb, const float* __restrict__ tproj,
             float* __restrict__ seqs)
{
    int bt = blockIdx.x, h = threadIdx.x;
    int id = logs[bt];
    float ts = tseq[bt];
    float acc = item_emb[(long)id * HH + h] * 11.313708498984761f; // sqrt(128)
    float p0 = TWO_PI * ts / 86400.f, p1 = TWO_PI * ts / 604800.f, p2 = TWO_PI * ts / 2592000.f;
    const float* tp = tproj + h * 6;
    acc += sinf(p0)*tp[0] + sinf(p1)*tp[1] + sinf(p2)*tp[2]
         + cosf(p0)*tp[3] + cosf(p1)*tp[4] + cosf(p2)*tp[5];
    seqs[(long)bt * HH + h] = (id == 0) ? 0.f : acc;
}

// phase-similarity bias: phi vectors are unit norm -> sim = sin*sin + cos*cos
__global__ __launch_bounds__(TT)
void pbias_k(const float* __restrict__ tseq, const float* __restrict__ lambdas,
             float* __restrict__ pb)
{
    int bt = blockIdx.x, s = threadIdx.x;
    int b = bt / TT;
    float tt = tseq[bt], ts = tseq[b*TT + s];
    const float P[3] = {86400.f, 604800.f, 2592000.f};
    float acc = 0.f;
#pragma unroll
    for (int c = 0; c < 3; ++c) {
        float pt = TWO_PI * tt / P[c];
        float ps = TWO_PI * ts / P[c];
        acc += lambdas[c] * (sinf(pt)*sinf(ps) + cosf(pt)*cosf(ps));
    }
    pb[(long)bt * TT + s] = acc;
}

// add on-the-fly gathered q.tK term, scale, phase bias, causal+pad mask
__global__ __launch_bounds__(TT)
void scorefix_k(float* __restrict__ S, const float* __restrict__ q,
                const int* __restrict__ tmat, const float* __restrict__ tKe,
                const float* __restrict__ pb, const int* __restrict__ logs)
{
    __shared__ float qs[DHH];
    int blk = blockIdx.x, k = threadIdx.x;
    int qrow = blk % TT, bh = blk / TT;
    int h = bh % NHH, b = bh / NHH;
    if (k < DHH) qs[k] = q[((long)bh * TT + qrow) * DHH + k];
    __syncthreads();
    int idx = tmat[((long)b * TT + qrow) * TT + k];
    const float* e = tKe + (long)idx * HH + h * DHH;
    float dot = 0.f;
#pragma unroll 8
    for (int d = 0; d < DHH; ++d) dot += qs[d] * e[d];
    float val = (S[(long)blk * TT + k] + dot) * 0.125f            // 1/sqrt(64)
              + pb[((long)b * TT + qrow) * TT + k];
    bool masked = (k > qrow) || (logs[b * TT + k] == 0);
    S[(long)blk * TT + k] = masked ? NEGV : val;
}

__global__ __launch_bounds__(TT)
void softmax_k(float* __restrict__ S)
{
    __shared__ float red[TT];
    long row = blockIdx.x;
    int k = threadIdx.x;
    float v = S[row*TT + k];
    red[k] = v; __syncthreads();
    for (int s = TT/2; s > 0; s >>= 1) { if (k < s) red[k] = fmaxf(red[k], red[k+s]); __syncthreads(); }
    float m = red[0]; __syncthreads();
    float e = __expf(v - m);
    red[k] = e; __syncthreads();
    for (int s = TT/2; s > 0; s >>= 1) { if (k < s) red[k] += red[k+s]; __syncthreads(); }
    S[row*TT + k] = e / red[0];
}

// A.tV gathered term + head merge + residual (Qn); A row staged via async LDS
__global__ __launch_bounds__(DHH)
void attnout_k(const float* __restrict__ A, const float* __restrict__ oh,
               const float* __restrict__ Qn, const int* __restrict__ tmat,
               const float* __restrict__ tVe, float* __restrict__ seqs)
{
    __shared__ float As[TT];
    int blk = blockIdx.x, d = threadIdx.x;
    int qrow = blk % TT, bh = blk / TT;
    int h = bh % NHH, b = bh / NHH;
#if USE_ASYNC_LDS
    {   // 64 lanes x 16B = 1KB row, ASYNCcnt-tracked, no VGPR round-trip
        const float* gsrc = A + (long)blk * TT + d * 4;
        __builtin_amdgcn_global_load_async_to_lds_b128(
            (g_v4i_p)(void*)(float*)(size_t)(const void*)gsrc,
            (s_v4i_p)(&As[d * 4]), 0, 0);
        __builtin_amdgcn_s_wait_asynccnt(0);
    }
    __syncthreads();
#else
    for (int k = d; k < TT; k += DHH) As[k] = A[(long)blk*TT + k];
    __syncthreads();
#endif
    const int* tm = tmat + ((long)b*TT + qrow)*TT;
    const float* base = tVe + h*DHH + d;
    float acc = 0.f;
    for (int k = 0; k < TT; ++k) acc += As[k] * base[(long)tm[k]*HH];
    long o = ((long)b*TT + qrow)*HH + h*DHH + d;
    seqs[o] = Qn[o] + oh[((long)bh*TT + qrow)*DHH + d] + acc;
}

__global__ __launch_bounds__(HH)
void mask_k(float* __restrict__ seqs, const int* __restrict__ logs)
{
    int bt = blockIdx.x;
    if (logs[bt] == 0) seqs[(long)bt*HH + threadIdx.x] = 0.f;
}

__global__ __launch_bounds__(HH)
void te_k(const float* __restrict__ ttimes, const float* __restrict__ tproj,
          float* __restrict__ te)
{
    int b = blockIdx.x, h = threadIdx.x;
    float ts = ttimes[b];
    float p0 = TWO_PI * ts / 86400.f, p1 = TWO_PI * ts / 604800.f, p2 = TWO_PI * ts / 2592000.f;
    const float* tp = tproj + h * 6;
    te[b*HH + h] = sinf(p0)*tp[0] + sinf(p1)*tp[1] + sinf(p2)*tp[2]
                 + cosf(p0)*tp[3] + cosf(p1)*tp[4] + cosf(p2)*tp[5];
}

__global__ __launch_bounds__(HH)
void hper_k(const float* __restrict__ te, const float* __restrict__ predW,
            float* __restrict__ hper)
{
    __shared__ float tr[HH];
    int b = blockIdx.x, h = threadIdx.x;
    tr[h] = te[b*HH + h];
    __syncthreads();
    const float* wr = predW + (long)h * HH;
    float acc = 0.f;
#pragma unroll 8
    for (int k = 0; k < HH; ++k) acc += tr[k] * wr[k];
    hper[b*HH + h] = acc;
}

__global__ __launch_bounds__(2*HH)
void cat_k(const float* __restrict__ logf, const float* __restrict__ hper,
           float* __restrict__ cat)
{
    int bt = blockIdx.x, j = threadIdx.x;
    int b = bt / TT;
    cat[(long)bt*2*HH + j] = (j < HH) ? logf[(long)bt*HH + j] : hper[b*HH + (j - HH)];
}

__global__ __launch_bounds__(HH)
void logits_k(const float* __restrict__ fused, const float* __restrict__ item_emb,
              const int* __restrict__ pos, const int* __restrict__ neg,
              float* __restrict__ out)
{
    __shared__ float red[HH];
    int bt = blockIdx.x, h = threadIdx.x;
    float f = fused[(long)bt*HH + h];
    red[h] = f * item_emb[(long)pos[bt]*HH + h];
    __syncthreads();
    for (int s = HH/2; s > 0; s >>= 1) { if (h < s) red[h] += red[h+s]; __syncthreads(); }
    if (h == 0) out[bt] = red[0];
    __syncthreads();
    red[h] = f * item_emb[(long)neg[bt]*HH + h];
    __syncthreads();
    for (int s = HH/2; s > 0; s >>= 1) { if (h < s) red[h] += red[h+s]; __syncthreads(); }
    if (h == 0) out[NTOK + bt] = red[0];
}

// ---------------------------------------------------------------------------
extern "C" void kernel_launch(void* const* d_in, const int* in_sizes, int n_in,
                              void* d_out, int out_size, void* d_ws, size_t ws_size,
                              hipStream_t stream)
{
    (void)in_sizes; (void)n_in; (void)out_size; (void)ws_size;
    const int*   logs   = (const int*)  d_in[0];
    const int*   tmat   = (const int*)  d_in[1];
    const float* tseq   = (const float*)d_in[2];
    const int*   pos    = (const int*)  d_in[3];
    const int*   neg    = (const int*)  d_in[4];
    const float* ttimes = (const float*)d_in[5];
    const float* itemE  = (const float*)d_in[6];
    const float* absK   = (const float*)d_in[7];
    const float* absV   = (const float*)d_in[8];
    const float* tKe    = (const float*)d_in[9];
    const float* tVe    = (const float*)d_in[10];
    const float* Wq = (const float*)d_in[11]; const float* bq = (const float*)d_in[12];
    const float* Wk = (const float*)d_in[13]; const float* bk = (const float*)d_in[14];
    const float* Wv = (const float*)d_in[15]; const float* bv = (const float*)d_in[16];
    const float* alg = (const float*)d_in[17]; const float* alb = (const float*)d_in[18];
    const float* flg = (const float*)d_in[19]; const float* flb = (const float*)d_in[20];
    const float* w1 = (const float*)d_in[21]; const float* b1 = (const float*)d_in[22];
    const float* w2 = (const float*)d_in[23]; const float* b2 = (const float*)d_in[24];
    const float* llg = (const float*)d_in[25]; const float* llb = (const float*)d_in[26];
    const float* tproj = (const float*)d_in[27];
    const float* lambdas = (const float*)d_in[28];
    const float* predW = (const float*)d_in[29];
    const float* fW1 = (const float*)d_in[30]; const float* fb1 = (const float*)d_in[31];
    const float* fW2 = (const float*)d_in[32]; const float* fb2 = (const float*)d_in[33];

    float* ws   = (float*)d_ws;
    float* seqs = ws;               // NTH
    float* Qn   = seqs + NTH;       // NTH
    float* qb   = Qn   + NTH;       // NTH
    float* kb   = qb   + NTH;       // NTH
    float* vb   = kb   + NTH;       // NTH
    float* oh   = vb   + NTH;       // NTH
    float* xb   = oh   + NTH;       // NTH
    float* h1   = xb   + NTH;       // NTH
    float* S    = h1   + NTH;       // B*NH*T*T
    float* pb   = S + (long)BB*NHH*TT*TT; // B*T*T
    // final-stage aliases (layer buffers are dead by then)
    float* logf   = qb;
    float* teb    = kb;
    float* hper   = kb + BB*HH;
    float* catb   = S;              // NTOK*2H <= B*NH*T*T
    float* fusedb = vb;

    const long sQKV = (long)TT*DHH;     // per-(b,h) stride of q/k/v
    const long sS   = (long)TT*TT;      // per-(b,h) stride of scores

    embed_k<<<NTOK, HH, 0, stream>>>(logs, tseq, itemE, tproj, seqs);
    pbias_k<<<NTOK, TT, 0, stream>>>(tseq, lambdas, pb);

    for (int l = 0; l < LL; ++l) {
        const long wo = (long)l*HH*HH;
        ln_k<<<NTOK, HH, 0, stream>>>(seqs, alg + l*HH, alb + l*HH, Qn);
        // q = LN(seqs)@Wq^T + bq               -> (B,NH,T,DH)
        wmma_gemm_k<1,0,1><<<dim3(HH/64, NTOK/16, 1), 32, 0, stream>>>(
            Qn, Wq + wo, bq + l*HH, nullptr, qb,
            NTOK, HH, HH, HH, HH, HH, 0, 0, 0);
        // k = seqs@Wk^T + bk + absK (folded)   -> (B,NH,T,DH)
        wmma_gemm_k<1,0,1><<<dim3(HH/64, NTOK/16, 1), 32, 0, stream>>>(
            seqs, Wk + wo, bk + l*HH, absK, kb,
            NTOK, HH, HH, HH, HH, HH, 0, 0, 0);
        // v = seqs@Wv^T + bv + absV (folded)
        wmma_gemm_k<1,0,1><<<dim3(HH/64, NTOK/16, 1), 32, 0, stream>>>(
            seqs, Wv + wo, bv + l*HH, absV, vb,
            NTOK, HH, HH, HH, HH, HH, 0, 0, 0);
        // S = q @ k'^T   (batched over b*NH)
        wmma_gemm_k<1,0,0><<<dim3(TT/64, TT/16, BB*NHH), 32, 0, stream>>>(
            qb, kb, nullptr, nullptr, S,
            TT, TT, DHH, DHH, DHH, TT, sQKV, sQKV, sS);
        scorefix_k<<<BB*NHH*TT, TT, 0, stream>>>(S, qb, tmat, tKe, pb, logs);
        softmax_k<<<BB*NHH*TT, TT, 0, stream>>>(S);
        // oh = A @ v'
        wmma_gemm_k<0,0,0><<<dim3(DHH/64, TT/16, BB*NHH), 32, 0, stream>>>(
            S, vb, nullptr, nullptr, oh,
            TT, DHH, TT, TT, DHH, DHH, sS, sQKV, sQKV);
        attnout_k<<<BB*NHH*TT, DHH, 0, stream>>>(S, oh, Qn, tmat, tVe, seqs);
        // FFN
        ln_k<<<NTOK, HH, 0, stream>>>(seqs, flg + l*HH, flb + l*HH, xb);
        wmma_gemm_k<1,1,0><<<dim3(HH/64, NTOK/16, 1), 32, 0, stream>>>(
            xb, w1 + wo, b1 + l*HH, nullptr, h1,
            NTOK, HH, HH, HH, HH, HH, 0, 0, 0);
        wmma_gemm_k<1,0,0><<<dim3(HH/64, NTOK/16, 1), 32, 0, stream>>>(
            h1, w2 + wo, b2 + l*HH, xb /*residual*/, seqs,
            NTOK, HH, HH, HH, HH, HH, 0, 0, 0);
        mask_k<<<NTOK, HH, 0, stream>>>(seqs, logs);
    }

    ln_k<<<NTOK, HH, 0, stream>>>(seqs, llg, llb, logf);
    te_k<<<BB, HH, 0, stream>>>(ttimes, tproj, teb);
    hper_k<<<BB, HH, 0, stream>>>(teb, predW, hper);
    cat_k<<<NTOK, 2*HH, 0, stream>>>(logf, hper, catb);
    // fused = gelu(cat @ fus_W1^T + fb1) @ fus_W2^T + fb2
    wmma_gemm_k<1,2,0><<<dim3(HH/64, NTOK/16, 1), 32, 0, stream>>>(
        catb, fW1, fb1, nullptr, h1,
        NTOK, HH, 2*HH, 2*HH, 2*HH, HH, 0, 0, 0);
    wmma_gemm_k<1,0,0><<<dim3(HH/64, NTOK/16, 1), 32, 0, stream>>>(
        h1, fW2, fb2, nullptr, fusedb,
        NTOK, HH, HH, HH, HH, HH, 0, 0, 0);
    logits_k<<<NTOK, HH, 0, stream>>>(fusedb, itemE, pos, neg, (float*)d_out);
}
